// SelfAttention_34016140984498
// MI455X (gfx1250) — matrix operand order repaired and checked by reading, not compiled, not run on hardware
//
#include <hip/hip_runtime.h>

// ---------- CDNA5 WMMA types ----------
typedef __attribute__((ext_vector_type(16))) _Float16 v16h;
typedef __attribute__((ext_vector_type(8)))  _Float16 v8h;
typedef __attribute__((ext_vector_type(8)))  float    v8f;

union F8 { v8f v; float f[8]; };

#define T_SEQ   2048
#define C_DIM   1024
#define C3      3072
#define H_NUM   16
#define D_HEAD  64
#define BT_ROWS 4096     // B*T
#define SCALE_Q 0.125f   // 64^-0.5
#define BK      64       // k-slab per LDS stage

__device__ __forceinline__ v8f wmma_f16(v16h a, v16h b, v8f c) {
  // D = A(16x32 f16) * B(32x16 f16) + C(16x16 f32)
  return __builtin_amdgcn_wmma_f32_16x16x32_f16(
      false, a, false, b, (short)0, c, false, false);
}

// A-matrix 16x32 fragment (row-major). rowp = &A[row*lda], row = lane&15.
// lane<16: K = k0+{0..15}; lane>=16: K = k0+{16..31}
__device__ __forceinline__ v16h a_frag(const _Float16* rowp, int k0, int lane) {
  const int kb = k0 + ((lane & 16) ? 16 : 0);
  union { v16h v; v8h h[2]; } u;
  u.h[0] = *(const v8h*)(rowp + kb);
  u.h[1] = *(const v8h*)(rowp + kb + 8);
  return u.v;
}

// B-matrix 32x16 fragment from column-contiguous layout. colp = &BT[col*ldt].
// lane<16: K = k0+{0..15}; lane>=16: K = k0+{16..31}. One 32B load.
__device__ __forceinline__ v16h b_frag(const _Float16* colp, int k0, int lane) {
  const int kb = k0 + ((lane & 16) ? 16 : 0);
  return *(const v16h*)(colp + kb);
}

// ---------- gfx1250 async global->LDS copy (16B per lane) ----------
__device__ __forceinline__ void async_cp16(_Float16* lptr, const _Float16* gptr) {
  const unsigned lds_off = (unsigned)(size_t)lptr;  // flat-shared addr[31:0] = LDS offset
  asm volatile("global_load_async_to_lds_b128 %0, %1, off"
               :: "v"(lds_off), "v"(gptr) : "memory");
}
__device__ __forceinline__ void async_wait0() {
  asm volatile("s_wait_asynccnt 0x0" ::: "memory");
}

// ---------- conversion kernels ----------
__global__ void cvt_f16_kernel(const float* __restrict__ in,
                               _Float16* __restrict__ out, int n) {
  int i = blockIdx.x * blockDim.x + threadIdx.x;
  const int stride = gridDim.x * blockDim.x;
  for (; i < n; i += stride) out[i] = (_Float16)in[i];
}

// in: [K][N] f32 -> out: [N][K] f16
__global__ void transpose_f16_kernel(const float* __restrict__ in,
                                     _Float16* __restrict__ out,
                                     int K, int N) {
  int i = blockIdx.x * blockDim.x + threadIdx.x;
  const int stride = gridDim.x * blockDim.x;
  const int total = K * N;
  for (; i < total; i += stride) {
    const int k = i / N, n = i - k * N;
    out[(size_t)n * K + k] = (_Float16)in[i];
  }
}

// ---------- QKV projection: 128x128 block tile, async double-buffered LDS ----------
// [4096,1024] x [1024,3072] + bias -> Q(scaled)/K as [B*H,T,64], V^T as [B*H,64,T]
__global__ __launch_bounds__(256)
void qkv_gemm_kernel(const _Float16* __restrict__ xh,
                     const _Float16* __restrict__ wT,   // [3072][1024]
                     const float* __restrict__ bias,
                     _Float16* __restrict__ qh,
                     _Float16* __restrict__ kh,
                     _Float16* __restrict__ vT) {
  __shared__ __align__(64) _Float16 Ab[2][128][BK];
  __shared__ __align__(64) _Float16 Bb[2][128][BK];
  const int tid   = threadIdx.x;
  const int lane  = tid & 31;
  const int wslot = tid >> 5;
  const int NB    = C3 / 128;                    // 24 col blocks
  const int mblk  = (blockIdx.x / NB) * 128;
  const int nblk  = (blockIdx.x % NB) * 128;
  const int mrow  = lane & 15;
  const int m_off = (wslot & 3) * 32;            // 4 row-waves
  const int n_off = (wslot >> 2) * 64;           // 2 col-waves

  // async staging: 128x64 halves per array = 1024 16B-chunks = 4 per thread
  int arow[4], aseg[4];
  const _Float16* gA[4];
  const _Float16* gB[4];
#pragma unroll
  for (int u = 0; u < 4; ++u) {
    const int idx = tid + 256 * u;
    arow[u] = idx >> 3;
    aseg[u] = (idx & 7) * 8;
    gA[u] = xh + (size_t)(mblk + arow[u]) * C_DIM + aseg[u];
    gB[u] = wT + (size_t)(nblk + arow[u]) * C_DIM + aseg[u];
  }
  auto issue = [&](int buf, int ko) {
#pragma unroll
    for (int u = 0; u < 4; ++u) {
      async_cp16(&Ab[buf][arow[u]][aseg[u]], gA[u] + ko);
      async_cp16(&Bb[buf][arow[u]][aseg[u]], gB[u] + ko);
    }
  };

  v8f acc[2][4];
#pragma unroll
  for (int rt = 0; rt < 2; ++rt)
#pragma unroll
    for (int ct = 0; ct < 4; ++ct) acc[rt][ct] = {};

  auto compute = [&](int buf) {
#pragma unroll
    for (int kc = 0; kc < BK; kc += 32) {
      const v16h a0 = a_frag(&Ab[buf][m_off + mrow][0],      kc, lane);
      const v16h a1 = a_frag(&Ab[buf][m_off + 16 + mrow][0], kc, lane);
#pragma unroll
      for (int ct = 0; ct < 4; ++ct) {
        const v16h b = b_frag(&Bb[buf][n_off + 16 * ct + mrow][0], kc, lane);
        acc[0][ct] = wmma_f16(a0, b, acc[0][ct]);
        acc[1][ct] = wmma_f16(a1, b, acc[1][ct]);
      }
    }
  };

  const int NKS = C_DIM / BK;                    // 16 stages
  issue(0, 0);
  for (int ks = 0; ks < NKS - 1; ++ks) {         // branch-free steady state
    async_wait0();
    __syncthreads();                             // stage ks resident
    issue((ks + 1) & 1, (ks + 1) * BK);          // always prefetch
    compute(ks & 1);
    __syncthreads();                             // reads done before overwrite
  }
  async_wait0();
  __syncthreads();
  compute((NKS - 1) & 1);                        // peeled last stage

  const int rbase = (lane & 16) ? 8 : 0;
  const int which = (nblk + n_off) >> 10;        // uniform per wave: 0=q,1=k,2=v
#pragma unroll
  for (int ct = 0; ct < 4; ++ct) {
    const int n = nblk + n_off + 16 * ct + mrow;
    const float bv = bias[n];
    const int c = n & (C_DIM - 1);
    const int h = c >> 6, d = c & 63;
#pragma unroll
    for (int rt = 0; rt < 2; ++rt) {
      F8 t; t.v = acc[rt][ct];
#pragma unroll
      for (int i = 0; i < 8; ++i) {
        const int r = mblk + m_off + 16 * rt + rbase + i;
        const int b = r >> 11, tt = r & (T_SEQ - 1);
        const size_t head = (size_t)(b * H_NUM + h);
        const float v = t.f[i] + bv;
        if (which == 0)      qh[(head * T_SEQ + tt) * D_HEAD + d] = (_Float16)(v * SCALE_Q);
        else if (which == 1) kh[(head * T_SEQ + tt) * D_HEAD + d] = (_Float16)v;
        else                 vT[(head * D_HEAD + d) * T_SEQ + tt] = (_Float16)v;
      }
    }
  }
}

// ---------- flash attention: 32 queries per wave (2 row-tiles) ----------
__global__ __launch_bounds__(256)
void attn_kernel(const _Float16* __restrict__ qh,
                 const _Float16* __restrict__ kh,
                 const _Float16* __restrict__ vT,
                 _Float16* __restrict__ attn) {
  __shared__ __align__(64) _Float16 ldsP[8][2][16][32];  // per-wave P tiles
  const int lane  = threadIdx.x & 31;
  const int wslot = threadIdx.x >> 5;
  const int wid   = blockIdx.x * 8 + wslot;      // 0..2047
  const int QT    = T_SEQ / 32;                  // 64
  const int head  = wid / QT;                    // b*16 + h
  const int q0    = (wid - head * QT) * 32;
  const _Float16* Q = qh + (size_t)head * T_SEQ * D_HEAD;
  const _Float16* K = kh + (size_t)head * T_SEQ * D_HEAD;
  const _Float16* V = vT + (size_t)head * D_HEAD * T_SEQ;
  const int mrow  = lane & 15;
  const int rbase = (lane & 16) ? 8 : 0;

  v16h aQ[2][2];
#pragma unroll
  for (int rt = 0; rt < 2; ++rt) {
    const _Float16* qrow = Q + (size_t)(q0 + 16 * rt + mrow) * D_HEAD;
    aQ[rt][0] = a_frag(qrow, 0, lane);
    aQ[rt][1] = a_frag(qrow, 32, lane);
  }

  F8 o[2][4];
  float ms[2][8], ls[2][8];
#pragma unroll
  for (int rt = 0; rt < 2; ++rt)
#pragma unroll
    for (int i = 0; i < 8; ++i) {
      o[rt][0].f[i] = o[rt][1].f[i] = o[rt][2].f[i] = o[rt][3].f[i] = 0.0f;
      ms[rt][i] = -1e30f; ls[rt][i] = 0.0f;
    }

  for (int j = 0; j < T_SEQ; j += 32) {
    // scores 32q x 32k: K fragments shared across both query row-tiles
    F8 s[2][2];
#pragma unroll
    for (int rt = 0; rt < 2; ++rt)
#pragma unroll
      for (int c = 0; c < 2; ++c) s[rt][c].v = {};
#pragma unroll
    for (int c = 0; c < 2; ++c) {
      const _Float16* kcol = K + (size_t)(j + 16 * c + mrow) * D_HEAD;
      const v16h b0 = b_frag(kcol, 0,  lane);
      const v16h b1 = b_frag(kcol, 32, lane);
      s[0][c].v = wmma_f16(aQ[0][0], b0, s[0][c].v);
      s[0][c].v = wmma_f16(aQ[0][1], b1, s[0][c].v);
      s[1][c].v = wmma_f16(aQ[1][0], b0, s[1][c].v);
      s[1][c].v = wmma_f16(aQ[1][1], b1, s[1][c].v);
    }

    // online softmax per row (row i spans the 16 lanes of this half)
#pragma unroll
    for (int rt = 0; rt < 2; ++rt) {
#pragma unroll
      for (int i = 0; i < 8; ++i) {
        const float v0 = s[rt][0].f[i], v1 = s[rt][1].f[i];
        float rm = fmaxf(v0, v1);
        rm = fmaxf(rm, __shfl_xor(rm, 1, 32));
        rm = fmaxf(rm, __shfl_xor(rm, 2, 32));
        rm = fmaxf(rm, __shfl_xor(rm, 4, 32));
        rm = fmaxf(rm, __shfl_xor(rm, 8, 32));
        const float mnew = fmaxf(ms[rt][i], rm);
        const float sc = __expf(ms[rt][i] - mnew);
        const float p0 = __expf(v0 - mnew);
        const float p1 = __expf(v1 - mnew);
        float rs = p0 + p1;
        rs += __shfl_xor(rs, 1, 32);
        rs += __shfl_xor(rs, 2, 32);
        rs += __shfl_xor(rs, 4, 32);
        rs += __shfl_xor(rs, 8, 32);
        ls[rt][i] = ls[rt][i] * sc + rs;
        ms[rt][i] = mnew;
        o[rt][0].f[i] *= sc; o[rt][1].f[i] *= sc;
        o[rt][2].f[i] *= sc; o[rt][3].f[i] *= sc;
        ldsP[wslot][rt][rbase + i][mrow]      = (_Float16)p0;
        ldsP[wslot][rt][rbase + i][16 + mrow] = (_Float16)p1;
      }
    }
    __syncthreads();                     // C-layout -> A-layout via LDS
    const v16h aP0 = a_frag(&ldsP[wslot][0][mrow][0], 0, lane);
    const v16h aP1 = a_frag(&ldsP[wslot][1][mrow][0], 0, lane);
    __syncthreads();

    // O += P(32x32) * V(32x64); V fragments shared across both row-tiles
#pragma unroll
    for (int c = 0; c < 4; ++c) {
      const v16h bv = b_frag(V + (size_t)(16 * c + mrow) * T_SEQ, j, lane);
      o[0][c].v = wmma_f16(aP0, bv, o[0][c].v);
      o[1][c].v = wmma_f16(aP1, bv, o[1][c].v);
    }
  }

  // normalize + scatter to concat layout [B,T,C] f16
  const int b = head >> 4, hh = head & 15;
#pragma unroll
  for (int rt = 0; rt < 2; ++rt) {
#pragma unroll
    for (int i = 0; i < 8; ++i) {
      const float inv = 1.0f / ls[rt][i];
      const int t = q0 + 16 * rt + rbase + i;
      const size_t rowoff = ((size_t)(b * T_SEQ + t)) * C_DIM + hh * D_HEAD;
      attn[rowoff + 0  + mrow] = (_Float16)(o[rt][0].f[i] * inv);
      attn[rowoff + 16 + mrow] = (_Float16)(o[rt][1].f[i] * inv);
      attn[rowoff + 32 + mrow] = (_Float16)(o[rt][2].f[i] * inv);
      attn[rowoff + 48 + mrow] = (_Float16)(o[rt][3].f[i] * inv);
    }
  }
}

// ---------- output projection: 128x128 block tile, async double-buffered LDS ----------
__global__ __launch_bounds__(256)
void out_gemm_kernel(const _Float16* __restrict__ ah,
                     const _Float16* __restrict__ wT,   // [1024][1024]
                     const float* __restrict__ bias,
                     float* __restrict__ out) {
  __shared__ __align__(64) _Float16 Ab[2][128][BK];
  __shared__ __align__(64) _Float16 Bb[2][128][BK];
  const int tid   = threadIdx.x;
  const int lane  = tid & 31;
  const int wslot = tid >> 5;
  const int NB    = C_DIM / 128;                 // 8
  const int mblk  = (blockIdx.x / NB) * 128;
  const int nblk  = (blockIdx.x % NB) * 128;
  const int mrow  = lane & 15;
  const int m_off = (wslot & 3) * 32;
  const int n_off = (wslot >> 2) * 64;

  int arow[4], aseg[4];
  const _Float16* gA[4];
  const _Float16* gB[4];
#pragma unroll
  for (int u = 0; u < 4; ++u) {
    const int idx = tid + 256 * u;
    arow[u] = idx >> 3;
    aseg[u] = (idx & 7) * 8;
    gA[u] = ah + (size_t)(mblk + arow[u]) * C_DIM + aseg[u];
    gB[u] = wT + (size_t)(nblk + arow[u]) * C_DIM + aseg[u];
  }
  auto issue = [&](int buf, int ko) {
#pragma unroll
    for (int u = 0; u < 4; ++u) {
      async_cp16(&Ab[buf][arow[u]][aseg[u]], gA[u] + ko);
      async_cp16(&Bb[buf][arow[u]][aseg[u]], gB[u] + ko);
    }
  };

  v8f acc[2][4];
#pragma unroll
  for (int rt = 0; rt < 2; ++rt)
#pragma unroll
    for (int ct = 0; ct < 4; ++ct) acc[rt][ct] = {};

  auto compute = [&](int buf) {
#pragma unroll
    for (int kc = 0; kc < BK; kc += 32) {
      const v16h a0 = a_frag(&Ab[buf][m_off + mrow][0],      kc, lane);
      const v16h a1 = a_frag(&Ab[buf][m_off + 16 + mrow][0], kc, lane);
#pragma unroll
      for (int ct = 0; ct < 4; ++ct) {
        const v16h b = b_frag(&Bb[buf][n_off + 16 * ct + mrow][0], kc, lane);
        acc[0][ct] = wmma_f16(a0, b, acc[0][ct]);
        acc[1][ct] = wmma_f16(a1, b, acc[1][ct]);
      }
    }
  };

  const int NKS = C_DIM / BK;                    // 16
  issue(0, 0);
  for (int ks = 0; ks < NKS - 1; ++ks) {
    async_wait0();
    __syncthreads();
    issue((ks + 1) & 1, (ks + 1) * BK);
    compute(ks & 1);
    __syncthreads();
  }
  async_wait0();
  __syncthreads();
  compute((NKS - 1) & 1);

  const int rbase = (lane & 16) ? 8 : 0;
#pragma unroll
  for (int ct = 0; ct < 4; ++ct) {
    const int n = nblk + n_off + 16 * ct + mrow;
    const float bv = bias[n];
#pragma unroll
    for (int rt = 0; rt < 2; ++rt) {
      F8 t; t.v = acc[rt][ct];
#pragma unroll
      for (int i = 0; i < 8; ++i)
        out[(size_t)(mblk + m_off + 16 * rt + rbase + i) * C_DIM + n] =
            t.f[i] + bv;
    }
  }
}

// ---------- launch ----------
extern "C" void kernel_launch(void* const* d_in, const int* in_sizes, int n_in,
                              void* d_out, int out_size, void* d_ws, size_t ws_size,
                              hipStream_t stream) {
  const float* x      = (const float*)d_in[0];
  const float* w_qkv  = (const float*)d_in[1];
  const float* b_qkv  = (const float*)d_in[2];
  const float* w_out  = (const float*)d_in[3];
  const float* b_out  = (const float*)d_in[4];
  float* out = (float*)d_out;

  char* ws = (char*)d_ws;
  _Float16* x_h   = (_Float16*)ws;  ws += (size_t)BT_ROWS * C_DIM * 2;
  _Float16* wqkvT = (_Float16*)ws;  ws += (size_t)C3 * C_DIM * 2;
  _Float16* woT   = (_Float16*)ws;  ws += (size_t)C_DIM * C_DIM * 2;
  _Float16* qh    = (_Float16*)ws;  ws += (size_t)BT_ROWS * C_DIM * 2;
  _Float16* kh    = (_Float16*)ws;  ws += (size_t)BT_ROWS * C_DIM * 2;
  _Float16* vT    = (_Float16*)ws;  ws += (size_t)BT_ROWS * C_DIM * 2;
  _Float16* attn  = (_Float16*)ws;  ws += (size_t)BT_ROWS * C_DIM * 2;

  cvt_f16_kernel<<<2048, 256, 0, stream>>>(x, x_h, BT_ROWS * C_DIM);
  transpose_f16_kernel<<<2048, 256, 0, stream>>>(w_qkv, wqkvT, C_DIM, C3);
  transpose_f16_kernel<<<1024, 256, 0, stream>>>(w_out, woT, C_DIM, C_DIM);

  // 32 x 24 block tiles of 128x128
  qkv_gemm_kernel<<<768, 256, 0, stream>>>(x_h, wqkvT, b_qkv, qh, kh, vT);
  // 2*16*64 query tiles of 32 rows, 8 waves/block
  attn_kernel<<<256, 256, 0, stream>>>(qh, kh, vT, attn);
  // 32 x 8 block tiles of 128x128
  out_gemm_kernel<<<256, 256, 0, stream>>>(attn, woT, b_out, out);
}
